// MultiHeadAttention_12438225289951
// MI455X (gfx1250) — compile-verified, matrix operand order checked by reading, not run on hardware
//
#include <hip/hip_runtime.h>
#include <hip/hip_bf16.h>

// ---------------------------------------------------------------------------
// Problem constants (reference: B=8, S=2048, H=512, NH=8, DH=64)
// ---------------------------------------------------------------------------
static constexpr int cB  = 8;
static constexpr int cS  = 2048;
static constexpr int cH  = 512;
static constexpr int cNH = 8;
static constexpr int cDH = 64;
static constexpr int cBS = cB * cS;

typedef _Float16 h16;
typedef __attribute__((ext_vector_type(16))) _Float16 v16h;
typedef __attribute__((ext_vector_type(8)))  float    v8f;

union FragU { unsigned int u[8]; v16h v; };

// ---------------------------------------------------------------------------
// WMMA helpers (CDNA5 wave32, V_WMMA_F32_16X16X32_F16)
// 16-bit A-matrix 16x32 layout (ISA 7.12.2): lane L holds row M = L%16;
// half = L/16 selects k-groups; VGPR v holds the dword (k-pair) at index
//   v<4 ? v + half*4 : (v+4) + half*4     within the row's 32-half slice.
// B fragments mirror this, reading from a [N,K] row-major (pre-transposed)
// array so every fragment load is a contiguous dword read.
// ---------------------------------------------------------------------------
__device__ __forceinline__ v16h load_frag(const h16* row_base, int half) {
  const unsigned int* p = (const unsigned int*)row_base;
  FragU f;
#pragma unroll
  for (int v = 0; v < 8; ++v) {
    int idx = (v < 4) ? (v + half * 4) : (v + 4 + half * 4);
    f.u[v] = p[idx];
  }
  return f.v;
}

__device__ __forceinline__ v8f wmma32(v16h a, v16h b, v8f c) {
  return __builtin_amdgcn_wmma_f32_16x16x32_f16(false, a, false, b,
                                                (short)0, c, false, false);
}

// ---------------------------------------------------------------------------
// CDNA5 async Global->LDS copy (ASYNCcnt path, ISA ch.8 / 15.18.3 op 98).
// VDST VGPR carries the LDS byte address = low 32 bits of the generic
// shared-aperture address; the DMA writes LDS while the wave keeps issuing.
// ---------------------------------------------------------------------------
__device__ __forceinline__ void async_copy_b128(const h16* g, h16* l) {
  unsigned int laddr = (unsigned int)(unsigned long long)l;
  asm volatile("global_load_async_to_lds_b128 %0, %1, off"
               :: "v"(laddr), "v"(g) : "memory");
}
__device__ __forceinline__ void wait_async0() {
  asm volatile("s_wait_asynccnt 0x0" ::: "memory");
}
__device__ __forceinline__ void wait_async1() {   // newest copy may stay in flight
  asm volatile("s_wait_asynccnt 0x1" ::: "memory");
}

// ---------------------------------------------------------------------------
// Elementwise / prep kernels
// ---------------------------------------------------------------------------
__global__ void k_f32_to_h16(const float* __restrict__ in,
                             h16* __restrict__ out, size_t n) {
  size_t i = (size_t)blockIdx.x * blockDim.x + threadIdx.x;
  for (; i < n; i += (size_t)gridDim.x * blockDim.x)
    out[i] = (h16)in[i];
}

// Wt[n*H + k] = (h16) W[k*H + n]
__global__ void k_transpose_w(const float* __restrict__ W,
                              h16* __restrict__ Wt) {
  size_t i = (size_t)blockIdx.x * blockDim.x + threadIdx.x;
  if (i >= (size_t)cH * cH) return;
  int n = (int)(i / cH);
  int k = (int)(i % cH);
  Wt[i] = (h16)W[(size_t)k * cH + n];
}

// per-(b,h) L2 norm of x_m over the sequence axis (reference dim=1)
__global__ void k_colnorm(const float* __restrict__ xm,
                          float* __restrict__ cn) {
  int blk = blockIdx.x;                   // B * (H/256) = 16 blocks
  int b = blk >> 1;
  int h = ((blk & 1) << 8) + threadIdx.x;
  float s = 0.f;
  for (int ss = 0; ss < cS; ++ss) {
    float v = xm[((size_t)b * cS + ss) * cH + h];
    s += v * v;
  }
  cn[b * cH + h] = fmaxf(sqrtf(s), 1e-12f);
}

__global__ void k_make_z(const float* __restrict__ xm,
                         const float* __restrict__ cn,
                         h16* __restrict__ zb, size_t n) {
  size_t i = (size_t)blockIdx.x * blockDim.x + threadIdx.x;
  for (; i < n; i += (size_t)gridDim.x * blockDim.x) {
    int b = (int)(i / ((size_t)cS * cH));
    int h = (int)(i % cH);
    zb[i] = (h16)(xm[i] / cn[b * cH + h]);
  }
}

// row LayerNorm over H=512, one block per row
__global__ void k_layernorm(const float* __restrict__ y,
                            const float* __restrict__ lw,
                            const float* __restrict__ lb,
                            float* __restrict__ out) {
  __shared__ float sbuf[256];
  int row = blockIdx.x;
  int tid = threadIdx.x;
  const float* yr = y + (size_t)row * cH;
  float v0 = yr[tid], v1 = yr[tid + 256];
  sbuf[tid] = v0 + v1;
  __syncthreads();
  for (int st = 128; st > 0; st >>= 1) {
    if (tid < st) sbuf[tid] += sbuf[tid + st];
    __syncthreads();
  }
  float mu = sbuf[0] * (1.f / cH);
  __syncthreads();
  float d0 = v0 - mu, d1 = v1 - mu;
  sbuf[tid] = d0 * d0 + d1 * d1;
  __syncthreads();
  for (int st = 128; st > 0; st >>= 1) {
    if (tid < st) sbuf[tid] += sbuf[tid + st];
    __syncthreads();
  }
  float rstd = rsqrtf(sbuf[0] * (1.f / cH) + 1e-12f);
  float* orow = out + (size_t)row * cH;
  orow[tid]       = d0 * rstd * lw[tid]       + lb[tid];
  orow[tid + 256] = d1 * rstd * lw[tid + 256] + lb[tid + 256];
}

// ---------------------------------------------------------------------------
// WMMA GEMM: C[M,N] = A[M,K](h16) @ Bt[N,K](h16)^T  (+ epilogue)
//   mode 0: out_h = acc + bias                            (Q/K/V proj)
//   mode 1: out_f = 0.5*(m1+m2)*x*sigmoid(acc + bias)     (gating -> x_m)
//   mode 2: out_h = ctx * silu(acc + bias)                (gate branch)
//   mode 3: out_f = acc + bias + x                        (down + residual)
//   mode 4: out_h = acc < 0 ? 0.01 : acc                  (item_sim, no bias)
// Block = 256 threads = 8 waves sharing one 64-col B strip.  The B k-tile is
// double-buffered in LDS via async Global->LDS DMA: copy of tile i+1 overlaps
// the 32 WMMA of tile i (s_wait_asynccnt 1 keeps the newest copy in flight).
// ---------------------------------------------------------------------------
__global__ void k_gemm16(const h16* __restrict__ A,
                         const h16* __restrict__ Bt,
                         const float* __restrict__ bias,
                         const float* __restrict__ auxf,
                         const h16* __restrict__ auxh,
                         const float* __restrict__ m1,
                         const float* __restrict__ m2,
                         h16* __restrict__ outh,
                         float* __restrict__ outf,
                         int N, int K, int mode) {
  __shared__ __align__(16) h16 btile[2][64 * 32];
  int tid  = threadIdx.x;
  int lane = tid & 31;
  int w    = tid >> 5;
  int half = lane >> 4;
  int r16  = lane & 15;
  int m0 = blockIdx.y * 128 + w * 16;
  int n0 = blockIdx.x * 64;

  // async staging map: thread t copies 8 halves of B row (t>>2), chunk (t&3)
  int crow   = tid >> 2;
  int cchunk = (tid & 3) * 8;
  const h16* bsrc = Bt + (size_t)(n0 + crow) * K + cchunk;
  const h16* arow = A + (size_t)(m0 + r16) * K;

  const int nk = K / 32;
  async_copy_b128(bsrc, &btile[0][crow * 32 + cchunk]);   // preload tile 0

  v8f acc[4] = {};
  for (int i = 0; i < nk; ++i) {
    if (i + 1 < nk) {
      async_copy_b128(bsrc + (size_t)(i + 1) * 32,
                      &btile[(i + 1) & 1][crow * 32 + cchunk]);
      wait_async1();                      // tile i done, tile i+1 in flight
    } else {
      wait_async0();
    }
    __syncthreads();
    v16h a = load_frag(arow + i * 32, half);
    if (i + 1 < nk) __builtin_prefetch(arow + (i + 1) * 32, 0, 0);
    const h16* bt = &btile[i & 1][0];
#pragma unroll
    for (int j = 0; j < 4; ++j) {
      v16h bf = load_frag(bt + (j * 16 + r16) * 32, half);
      acc[j] = wmma32(a, bf, acc[j]);
    }
    __syncthreads();                      // readers done before next overwrite
  }

#pragma unroll
  for (int j = 0; j < 4; ++j) {
    int col = n0 + j * 16 + r16;
#pragma unroll
    for (int r = 0; r < 8; ++r) {
      int row = m0 + r + half * 8;
      size_t oi = (size_t)row * N + col;
      float v = acc[j][r];
      if (mode != 4) v += bias[col];
      if (mode == 0) {
        outh[oi] = (h16)v;
      } else if (mode == 1) {
        float sg = 1.f / (1.f + __expf(-v));
        outf[oi] = 0.5f * (m1[col] + m2[col]) * auxf[oi] * sg;
      } else if (mode == 2) {
        float g = v / (1.f + __expf(-v));       // silu
        outh[oi] = (h16)((float)auxh[oi] * g);
      } else if (mode == 3) {
        outf[oi] = v + auxf[oi];
      } else {                                  // item_sim clamp
        outh[oi] = (h16)((v < 0.f) ? 0.01f : v);
      }
    }
  }
}

// ---------------------------------------------------------------------------
// Flash attention: block = 256 threads = 8 waves; each wave owns 16 q-rows.
// Per 32-key tile: K tile staged in LDS by async DMA (shared by all 8 waves),
// V tile transposed into LDS; 4 WMMA (q.kT) + online softmax + 4 WMMA (P.V).
// item_sim tile (fp16, L2-resident) is mixed into the logits pre-softmax.
// ---------------------------------------------------------------------------
__global__ void k_flash_attn(const h16* __restrict__ qb,
                             const h16* __restrict__ kb,
                             const h16* __restrict__ vb,
                             const h16* __restrict__ simb,
                             const float* __restrict__ mix,
                             h16* __restrict__ ctxb) {
  __shared__ __align__(16) h16 ktile[32 * 64];   // raw K tile [key][dh]
  __shared__ __align__(16) h16 vT[64][34];       // transposed V tile (+pad)
  __shared__ __align__(16) h16 pbuf[8][16 * 32]; // per-wave P scratch

  int tid  = threadIdx.x;
  int lane = tid & 31;
  int w    = tid >> 5;
  int half = lane >> 4;
  int r16  = lane & 15;
  int b    = blockIdx.z;
  int head = blockIdx.y;
  int q0   = blockIdx.x * 128 + w * 16;

  // softmax(mix_weights)
  float a0 = mix[0], a1 = mix[1];
  float mxw = fmaxf(a0, a1);
  float e0 = __expf(a0 - mxw), e1 = __expf(a1 - mxw);
  float w0 = e0 / (e0 + e1), w1 = e1 / (e0 + e1);
  const float rscale = 0.125f;                   // 1/sqrt(DH)

  // persistent q fragments (DH=64 -> 2 frags)
  v16h qf0, qf1;
  {
    const h16* qrow = qb + ((size_t)b * cS + q0 + r16) * cH + head * cDH;
    qf0 = load_frag(qrow, half);
    qf1 = load_frag(qrow + 32, half);
  }

  // async K staging map: thread t copies 8 halves of key row (t>>3)
  int kkrow = tid >> 3;
  int kkoff = (tid & 7) * 8;

  float mrow[8], lrow[8];
  v8f o[4] = {};
#pragma unroll
  for (int r = 0; r < 8; ++r) { mrow[r] = -1e30f; lrow[r] = 0.f; }

  for (int kt = 0; kt < cS / 32; ++kt) {
    __syncthreads();   // prior iteration's LDS readers done
    async_copy_b128(
        kb + ((size_t)b * cS + kt * 32 + kkrow) * cH + head * cDH + kkoff,
        &ktile[kkrow * 64 + kkoff]);
    // transposed V tile: vT[d][kk] = V[kt*32+kk][d]
    for (int idx = tid; idx < 32 * 64; idx += 256) {
      int kk = idx >> 6, d = idx & 63;
      vT[d][kk] = vb[((size_t)b * cS + kt * 32 + kk) * cH + head * cDH + d];
    }
    wait_async0();
    __syncthreads();

    // scores for two 16-key halves (K frags straight out of LDS)
    v8f sc[2];
#pragma unroll
    for (int nh2 = 0; nh2 < 2; ++nh2) {
      v8f accS = {};
      const h16* krow = &ktile[(nh2 * 16 + r16) * 64];
      accS = wmma32(qf0, load_frag(krow, half), accS);
      accS = wmma32(qf1, load_frag(krow + 32, half), accS);
#pragma unroll
      for (int r = 0; r < 8; ++r) {
        size_t si = ((size_t)b * cS + q0 + r + half * 8) * cS +
                    kt * 32 + nh2 * 16 + r16;
        float sv = (float)simb[si];        // pre-clamped, L2-hot
        accS[r] = (w0 * accS[r] + w1 * sv) * rscale;
      }
      sc[nh2] = accS;
    }

    // online softmax (row = one 16-lane half-group; shfl_xor masks 1..8)
#pragma unroll
    for (int r = 0; r < 8; ++r) {
      float t = fmaxf(sc[0][r], sc[1][r]);
      for (int m = 1; m <= 8; m <<= 1) t = fmaxf(t, __shfl_xor(t, m, 32));
      float mnew = fmaxf(mrow[r], t);
      float p0 = __expf(sc[0][r] - mnew);
      float p1 = __expf(sc[1][r] - mnew);
      float rs = p0 + p1;
      for (int m = 1; m <= 8; m <<= 1) rs += __shfl_xor(rs, m, 32);
      float scale = __expf(mrow[r] - mnew);
      lrow[r] = lrow[r] * scale + rs;
      mrow[r] = mnew;
#pragma unroll
      for (int j = 0; j < 4; ++j) o[j][r] *= scale;
      int rr = r + half * 8;
      pbuf[w][rr * 32 + r16]      = (h16)p0;
      pbuf[w][rr * 32 + 16 + r16] = (h16)p1;
    }

    // P @ V : reload P in A-layout from wave-private LDS
    v16h pf = load_frag(&pbuf[w][r16 * 32], half);
#pragma unroll
    for (int j = 0; j < 4; ++j) {
      v16h vf = load_frag(&vT[j * 16 + r16][0], half);
      o[j] = wmma32(pf, vf, o[j]);
    }
  }

  // normalize and write ctx (fp16, consumed by the Wg/Wd WMMA GEMMs)
#pragma unroll
  for (int j = 0; j < 4; ++j)
#pragma unroll
    for (int r = 0; r < 8; ++r) {
      float val = o[j][r] / lrow[r];
      ctxb[((size_t)b * cS + q0 + r + half * 8) * cH + head * cDH +
           j * 16 + r16] = (h16)val;
    }
}

// ---------------------------------------------------------------------------
// Host-side orchestration
// ---------------------------------------------------------------------------
extern "C" void kernel_launch(void* const* d_in, const int* in_sizes, int n_in,
                              void* d_out, int out_size, void* d_ws,
                              size_t ws_size, hipStream_t stream) {
  (void)in_sizes; (void)n_in; (void)out_size; (void)ws_size;

  const float* x     = (const float*)d_in[0];
  const float* Wq    = (const float*)d_in[1];
  const float* bq    = (const float*)d_in[2];
  const float* Wk    = (const float*)d_in[3];
  const float* bk    = (const float*)d_in[4];
  const float* Wv    = (const float*)d_in[5];
  const float* bv    = (const float*)d_in[6];
  const float* Wg    = (const float*)d_in[7];
  const float* bg    = (const float*)d_in[8];
  const float* Wd    = (const float*)d_in[9];
  const float* bd    = (const float*)d_in[10];
  const float* ln_w  = (const float*)d_in[11];
  const float* ln_b  = (const float*)d_in[12];
  const float* m1    = (const float*)d_in[13];
  const float* m2    = (const float*)d_in[14];
  const float* Wgate = (const float*)d_in[15];
  const float* bgate = (const float*)d_in[16];
  const float* mix   = (const float*)d_in[17];
  float* out = (float*)d_out;

  // workspace carve-up (256B aligned)
  size_t off = 0;
  auto alloc = [&](size_t bytes) -> void* {
    void* p = (char*)d_ws + off;
    off += (bytes + 255) & ~(size_t)255;
    return p;
  };
  h16* xb    = (h16*)alloc((size_t)cBS * cH * 2);
  h16* wt_g  = (h16*)alloc((size_t)cH * cH * 2);   // gating_weight^T
  h16* wt_q  = (h16*)alloc((size_t)cH * cH * 2);
  h16* wt_k  = (h16*)alloc((size_t)cH * cH * 2);
  h16* wt_v  = (h16*)alloc((size_t)cH * cH * 2);
  h16* wt_gg = (h16*)alloc((size_t)cH * cH * 2);   // Wg^T
  h16* wt_d  = (h16*)alloc((size_t)cH * cH * 2);   // Wd^T
  float* xm  = (float*)alloc((size_t)cBS * cH * 4);   // x_m, reused as y
  float* cn  = (float*)alloc((size_t)cB * cH * 4);
  h16* zb    = (h16*)alloc((size_t)cBS * cH * 2);
  h16* qb    = (h16*)alloc((size_t)cBS * cH * 2);
  h16* kb    = (h16*)alloc((size_t)cBS * cH * 2);
  h16* vb    = (h16*)alloc((size_t)cBS * cH * 2);
  h16* simb  = (h16*)alloc((size_t)cB * cS * cS * 2);
  h16* ctxb  = (h16*)alloc((size_t)cBS * cH * 2);
  h16* hb    = (h16*)alloc((size_t)cBS * cH * 2);
  float* yb  = xm;  // x_m dead after z; reuse for y = hs + x

  const size_t nElem = (size_t)cBS * cH;
  const dim3 gemmGrid(cH / 64, cBS / 128);      // 8 x 128 blocks of 256

  // 1) fp16 copies of x and the six (transposed) weight matrices
  k_f32_to_h16<<<4096, 256, 0, stream>>>(x, xb, nElem);
  k_transpose_w<<<(cH * cH + 255) / 256, 256, 0, stream>>>(Wgate, wt_g);
  k_transpose_w<<<(cH * cH + 255) / 256, 256, 0, stream>>>(Wq, wt_q);
  k_transpose_w<<<(cH * cH + 255) / 256, 256, 0, stream>>>(Wk, wt_k);
  k_transpose_w<<<(cH * cH + 255) / 256, 256, 0, stream>>>(Wv, wt_v);
  k_transpose_w<<<(cH * cH + 255) / 256, 256, 0, stream>>>(Wg, wt_gg);
  k_transpose_w<<<(cH * cH + 255) / 256, 256, 0, stream>>>(Wd, wt_d);

  // 2) gating GEMM -> x_m = 0.5*(m1+m2)*x*sigmoid(x@Wgate + bgate)
  k_gemm16<<<gemmGrid, 256, 0, stream>>>(xb, wt_g, bgate, x, nullptr, m1, m2,
                                         nullptr, xm, cH, cH, 1);
  // 3) sequence-axis normalization -> z (fp16)
  k_colnorm<<<cB * (cH / 256), 256, 0, stream>>>(xm, cn);
  k_make_z<<<4096, 256, 0, stream>>>(xm, cn, zb, nElem);

  // 4) Q/K/V projections
  k_gemm16<<<gemmGrid, 256, 0, stream>>>(xb, wt_q, bq, nullptr, nullptr,
                                         nullptr, nullptr, qb, nullptr, cH, cH, 0);
  k_gemm16<<<gemmGrid, 256, 0, stream>>>(xb, wt_k, bk, nullptr, nullptr,
                                         nullptr, nullptr, kb, nullptr, cH, cH, 0);
  k_gemm16<<<gemmGrid, 256, 0, stream>>>(xb, wt_v, bv, nullptr, nullptr,
                                         nullptr, nullptr, vb, nullptr, cH, cH, 0);

  // 5) item similarity per batch: sim = clamp(z @ z^T) (mode 4), fp16 [S,S]
  for (int b = 0; b < cB; ++b) {
    k_gemm16<<<dim3(cS / 64, cS / 128), 256, 0, stream>>>(
        zb + (size_t)b * cS * cH, zb + (size_t)b * cS * cH, nullptr,
        nullptr, nullptr, nullptr, nullptr,
        simb + (size_t)b * cS * cS, nullptr, cS, cH, 4);
  }

  // 6) flash attention with mixed scores
  k_flash_attn<<<dim3(cS / 128, cNH, cB), 256, 0, stream>>>(qb, kb, vb, simb,
                                                            mix, ctxb);

  // 7) gate branch: hb = ctx * silu(ctx@Wg + bg)
  k_gemm16<<<gemmGrid, 256, 0, stream>>>(ctxb, wt_gg, bg, nullptr, ctxb,
                                         nullptr, nullptr, hb, nullptr, cH, cH, 2);
  // 8) down proj + residual: y = hb@Wd + bd + x
  k_gemm16<<<gemmGrid, 256, 0, stream>>>(hb, wt_d, bd, x, nullptr,
                                         nullptr, nullptr, nullptr, yb, cH, cH, 3);

  // 9) LayerNorm -> output
  k_layernorm<<<cBS, 256, 0, stream>>>(yb, ln_w, ln_b, out);
}
